// LocalAttention_42872363549273
// MI455X (gfx1250) — matrix operand order repaired and checked by reading, not compiled, not run on hardware
//
#include <hip/hip_runtime.h>
#include <hip/hip_bf16.h>

typedef __attribute__((ext_vector_type(16))) _Float16 v16h;
typedef __attribute__((ext_vector_type(8)))  _Float16 v8h;
typedef __attribute__((ext_vector_type(8)))  float    v8f;
typedef __attribute__((ext_vector_type(4)))  float    v4f;
typedef __attribute__((ext_vector_type(4)))  unsigned int v4u;

#define DM    512
#define NH    8
#define HD    64
#define SEQ   4096
#define BATCH 2
#define WIN   65
#define HW    32
#define ROWS  8192        // BATCH*SEQ
#define KP_ROWS 4160      // SEQ + 64  (32 zero rows each side)
#define VP_COLS 4224      // SEQ + 128 (extra tail pad for last 32-k chunk)

// ---------- fragment helpers (ISA 7.12.2 layouts, wave32) ----------
union H16 { v16h v; v8h h[2]; };

__device__ __forceinline__ v16h frag_pair(const _Float16* p0, const _Float16* p1) {
    H16 u;
    u.h[0] = *(const v8h*)p0;
    u.h[1] = *(const v8h*)p1;
    return u.v;
}

__device__ __forceinline__ v8f wmma_f16(v16h a, v16h b, v8f c) {
    return __builtin_amdgcn_wmma_f32_16x16x32_f16(false, a, false, b, (short)0, c, false, false);
}

// ---------- fused prep: zero K/V pads + x->f16 + 4 weight transposes ----------
#define NZQ 1073152u   // b128 zero items  (KP bytes 8519680 + VP bytes 8650752)/16
#define NXC 524288u    // x-conversion items (8 floats each)
#define NWC 262144u    // per-weight transpose items (512*512)
// total threads = NZQ + NXC + 4*NWC = 2646016 = 10336 * 256

__global__ __launch_bounds__(256)
void prep_kernel(const float* __restrict__ x,
                 const float* __restrict__ Wq, const float* __restrict__ Wk,
                 const float* __restrict__ Wv, const float* __restrict__ Wo,
                 _Float16* __restrict__ xh,
                 _Float16* __restrict__ wtq, _Float16* __restrict__ wtk,
                 _Float16* __restrict__ wtv, _Float16* __restrict__ wto,
                 v4u* __restrict__ zbase)
{
    unsigned int i = blockIdx.x * 256 + threadIdx.x;
    if (i < NZQ) {
        zbase[i] = (v4u){0u, 0u, 0u, 0u};
    } else if (i < NZQ + NXC) {
        unsigned int j = i - NZQ;
        v4f a = *(const v4f*)(x + (size_t)j * 8);
        v4f b = *(const v4f*)(x + (size_t)j * 8 + 4);
        v8h o;
        o[0] = (_Float16)a[0]; o[1] = (_Float16)a[1]; o[2] = (_Float16)a[2]; o[3] = (_Float16)a[3];
        o[4] = (_Float16)b[0]; o[5] = (_Float16)b[1]; o[6] = (_Float16)b[2]; o[7] = (_Float16)b[3];
        *(v8h*)(xh + (size_t)j * 8) = o;
    } else {
        unsigned int j = i - NZQ - NXC;
        unsigned int w = j >> 18;                 // 0..3
        unsigned int idx = j & (NWC - 1);
        const float* W  = (w == 0) ? Wq  : (w == 1) ? Wk  : (w == 2) ? Wv  : Wo;
        _Float16*    Wt = (w == 0) ? wtq : (w == 1) ? wtk : (w == 2) ? wtv : wto;
        int n = idx >> 9, k = idx & (DM - 1);
        Wt[idx] = (_Float16)W[k * DM + n];        // Wt[n][k] = W[k][n]
    }
}

// ---------- WMMA GEMM tile body: 16(M) x 64(N) strip per wave ----------
// mode 0: Q -> f16 head-major, scaled 1/8
// mode 1: K -> f16 head-major, padded rows (+HW)
// mode 2: V -> f16 d-major transposed, padded cols (+HW)
// mode 3: O -> f32 row-major to d_out
__device__ __forceinline__
void gemm_tile(const _Float16* __restrict__ X, const _Float16* __restrict__ Wt,
               const float* __restrict__ bias, void* __restrict__ dst,
               int mode, int tile, int lane)
{
    int mt = tile >> 3;                  // 0..511
    int nb = tile & 7;                   // 0..7
    int m = lane & 15, half = lane >> 4;
    int r0   = mt * 16;
    int col0 = nb * 64;

    const _Float16* arow = X + (size_t)(r0 + m) * DM;
    const _Float16* bc0 = Wt + (size_t)(col0 + m) * DM;
    const _Float16* bc1 = bc0 + (size_t)16 * DM;
    const _Float16* bc2 = bc0 + (size_t)32 * DM;
    const _Float16* bc3 = bc0 + (size_t)48 * DM;

    // warm WGP$ ahead of the load clauses (global_prefetch_b8)
    __builtin_prefetch(arow, 0, 1);
    __builtin_prefetch(bc0, 0, 1);
    __builtin_prefetch(bc2, 0, 1);

    v8f acc0 = {}, acc1 = {}, acc2 = {}, acc3 = {};
#pragma unroll
    for (int kk = 0; kk < DM; kk += 32) {
        v16h a  = frag_pair(arow + kk + half * 16, arow + kk + half * 16 + 8);
        v16h b0 = frag_pair(bc0 + kk + half * 16, bc0 + kk + half * 16 + 8);
        v16h b1 = frag_pair(bc1 + kk + half * 16, bc1 + kk + half * 16 + 8);
        v16h b2 = frag_pair(bc2 + kk + half * 16, bc2 + kk + half * 16 + 8);
        v16h b3 = frag_pair(bc3 + kk + half * 16, bc3 + kk + half * 16 + 8);
        acc0 = wmma_f16(a, b0, acc0);
        acc1 = wmma_f16(a, b1, acc1);
        acc2 = wmma_f16(a, b2, acc2);
        acc3 = wmma_f16(a, b3, acc3);
    }

#pragma unroll
    for (int sub = 0; sub < 4; ++sub) {
        v8f acc = (sub == 0) ? acc0 : (sub == 1) ? acc1 : (sub == 2) ? acc2 : acc3;
        int col = col0 + sub * 16 + m;
        float bv = bias[col];
        int h = col >> 6, dd = col & 63;
#pragma unroll
        for (int v = 0; v < 8; ++v) {
            int row  = r0 + v + 8 * half;
            int bidx = row >> 12;            // / SEQ
            int n    = row & (SEQ - 1);
            int bh   = bidx * NH + h;
            float val = acc[v] + bv;
            if (mode == 0) {
                ((_Float16*)dst)[((size_t)bh * SEQ + n) * HD + dd] = (_Float16)(val * 0.125f);
            } else if (mode == 1) {
                ((_Float16*)dst)[((size_t)bh * KP_ROWS + (n + HW)) * HD + dd] = (_Float16)val;
            } else if (mode == 2) {
                ((_Float16*)dst)[((size_t)bh * HD + dd) * VP_COLS + (n + HW)] = (_Float16)val;
            } else {
                ((float*)dst)[(size_t)row * DM + col] = val;
            }
        }
    }
}

// Fused Q+K+V projections: 3072 blocks; mode is block-uniform (no divergence).
__global__ __launch_bounds__(128)
void gemm_qkv(const _Float16* __restrict__ X,
              const _Float16* __restrict__ wtq, const _Float16* __restrict__ wtk,
              const _Float16* __restrict__ wtv,
              const float* __restrict__ bq, const float* __restrict__ bk,
              const float* __restrict__ bv,
              _Float16* __restrict__ qh, _Float16* __restrict__ kp,
              _Float16* __restrict__ vp)
{
    int lane = threadIdx.x & 31;
    int mode = blockIdx.x >> 10;                        // 0,1,2 (1024 blocks each)
    int tile = (blockIdx.x & 1023) * 4 + (threadIdx.x >> 5);
    const _Float16* Wt  = (mode == 0) ? wtq : (mode == 1) ? wtk : wtv;
    const float*   bias = (mode == 0) ? bq  : (mode == 1) ? bk  : bv;
    void*           dst = (mode == 0) ? (void*)qh : (mode == 1) ? (void*)kp : (void*)vp;
    gemm_tile(X, Wt, bias, dst, mode, tile, lane);
}

// Output projection (fp32 result to d_out)
__global__ __launch_bounds__(128)
void gemm_o(const _Float16* __restrict__ X, const _Float16* __restrict__ wto,
            const float* __restrict__ bo, float* __restrict__ out)
{
    int lane = threadIdx.x & 31;
    int tile = blockIdx.x * 4 + (threadIdx.x >> 5);
    gemm_tile(X, wto, bo, (void*)out, 3, tile, lane);
}

// ---------- banded attention: one 16-row tile per wave ----------
__global__ __launch_bounds__(128)
void attn_wmma(const _Float16* __restrict__ qh, const _Float16* __restrict__ kp,
               const _Float16* __restrict__ vp, _Float16* __restrict__ ctx,
               float* __restrict__ attn_out)
{
    __shared__ __align__(16) float    sS[4][16 * 80];
    __shared__ __align__(16) _Float16 sP[4][16 * 96];

    int lane = threadIdx.x & 31;
    int wv   = threadIdx.x >> 5;
    int tile = blockIdx.x * 4 + wv;      // 4096 tiles = 16 bh x 256
    int bh = tile >> 8;
    int n0 = (tile & 255) * 16;
    int m = lane & 15, half = lane >> 4;

    // Q A-fragments (K-dim 64 -> two chunks of 32); Q already scaled by 1/sqrt(d)
    const _Float16* qrow = qh + ((size_t)bh * SEQ + n0 + m) * HD;
    v16h aq0 = frag_pair(qrow + half * 16,      qrow + half * 16 + 8);
    v16h aq1 = frag_pair(qrow + 32 + half * 16, qrow + 32 + half * 16 + 8);

    // scores over union window: keys g = n0-32 .. n0+47 (padded index gp = n0 + 0..79)
    const _Float16* kbase = kp + (size_t)bh * KP_ROWS * HD;
#pragma unroll
    for (int j = 0; j < 5; ++j) {
        const _Float16* kr = kbase + (size_t)(n0 + j * 16 + m) * HD;  // column = key
        v16h b0 = frag_pair(kr + half * 16,      kr + half * 16 + 8);
        v16h b1 = frag_pair(kr + 32 + half * 16, kr + 32 + half * 16 + 8);
        v8f s = {};
        s = wmma_f16(aq0, b0, s);
        s = wmma_f16(aq1, b1, s);
#pragma unroll
        for (int v = 0; v < 8; ++v)
            sS[wv][(v + 8 * half) * 80 + j * 16 + m] = s[v];
    }
    __syncthreads();

    // per-row softmax over the contiguous 65-wide valid band [r, r+64]
    if (lane < 16) {
        int r = lane;
        const float* srow = &sS[wv][r * 80];
        float mx = -1e30f;
        for (int w = 0; w <= 64; ++w) mx = fmaxf(mx, srow[r + w]);
        float sum = 0.f;
        for (int w = 0; w <= 64; ++w) sum += __expf(srow[r + w] - mx);
        float inv = 1.f / sum;
        float* aout = attn_out + ((size_t)bh * SEQ + (n0 + r)) * WIN;
        _Float16* prow = &sP[wv][r * 96];
        for (int ct = 0; ct < 96; ++ct) {
            float pv = 0.f;
            if (ct >= r && ct <= r + 64) {
                pv = __expf(srow[ct] - mx) * inv;
                aout[ct - r] = pv;
            }
            prow[ct] = (_Float16)pv;
        }
    }
    __syncthreads();

    // ctx(16x64) = P(16x96, tail zero) @ Vwin(96x64)  -- V stored d-major, padded
    const _Float16* vbase = vp + (size_t)bh * HD * VP_COLS;
    v8f c0 = {}, c1 = {}, c2 = {}, c3 = {};
#pragma unroll
    for (int kc = 0; kc < 3; ++kc) {
        const _Float16* pr = &sP[wv][m * 96 + kc * 32];
        v16h ap = frag_pair(pr + half * 16, pr + half * 16 + 8);
#pragma unroll
        for (int nbk = 0; nbk < 4; ++nbk) {
            const _Float16* vr = vbase + (size_t)(nbk * 16 + m) * VP_COLS + n0 + kc * 32;
            v16h bfrag = frag_pair(vr + half * 16, vr + half * 16 + 8);
            v8f cc = (nbk == 0) ? c0 : (nbk == 1) ? c1 : (nbk == 2) ? c2 : c3;
            cc = wmma_f16(ap, bfrag, cc);
            if      (nbk == 0) c0 = cc;
            else if (nbk == 1) c1 = cc;
            else if (nbk == 2) c2 = cc;
            else               c3 = cc;
        }
    }

    // store ctx f16 as [B*SEQ][512] row-major (un-permuted) for the Wo GEMM
    int bidx = bh >> 3, h = bh & 7;
#pragma unroll
    for (int nbk = 0; nbk < 4; ++nbk) {
        v8f cc = (nbk == 0) ? c0 : (nbk == 1) ? c1 : (nbk == 2) ? c2 : c3;
        int colg = h * 64 + nbk * 16 + m;
#pragma unroll
        for (int v = 0; v < 8; ++v) {
            int n = n0 + v + 8 * half;
            ctx[((size_t)bidx * SEQ + n) * DM + colg] = (_Float16)cc[v];
        }
    }
}

// ---------- workspace layout (bytes) ----------
#define OFF_XH   ((size_t)0)                       // 8192*512*2      = 8388608
#define OFF_WTQ  ((size_t)8388608)                 // 512*512*2 each
#define OFF_WTK  ((size_t)8912896)
#define OFF_WTV  ((size_t)9437184)
#define OFF_WTO  ((size_t)9961472)
#define OFF_QH   ((size_t)10485760)                // 16*4096*64*2    = 8388608
#define OFF_KP   ((size_t)18874368)                // 16*4160*64*2    = 8519680
#define OFF_VP   ((size_t)27394048)                // 16*64*4224*2    = 8650752
#define OFF_CTX  ((size_t)36044800)                // 8192*512*2      = 8388608
// total = 44433408 bytes

extern "C" void kernel_launch(void* const* d_in, const int* in_sizes, int n_in,
                              void* d_out, int out_size, void* d_ws, size_t ws_size,
                              hipStream_t stream) {
    const float* x  = (const float*)d_in[0];
    const float* Wq = (const float*)d_in[1];
    const float* Wk = (const float*)d_in[2];
    const float* Wv = (const float*)d_in[3];
    const float* Wo = (const float*)d_in[4];
    const float* bq = (const float*)d_in[5];
    const float* bk = (const float*)d_in[6];
    const float* bv = (const float*)d_in[7];
    const float* bo = (const float*)d_in[8];

    char* ws = (char*)d_ws;
    _Float16* xh   = (_Float16*)(ws + OFF_XH);
    _Float16* wtq  = (_Float16*)(ws + OFF_WTQ);
    _Float16* wtk  = (_Float16*)(ws + OFF_WTK);
    _Float16* wtv  = (_Float16*)(ws + OFF_WTV);
    _Float16* wto  = (_Float16*)(ws + OFF_WTO);
    _Float16* q_h  = (_Float16*)(ws + OFF_QH);
    _Float16* k_p  = (_Float16*)(ws + OFF_KP);
    _Float16* v_p  = (_Float16*)(ws + OFF_VP);
    _Float16* ctxh = (_Float16*)(ws + OFF_CTX);

    float* out      = (float*)d_out;                       // 2*4096*512 floats
    float* attn_out = (float*)d_out + (size_t)ROWS * DM;   // 2*8*4096*65 floats

    // 1) fused prep: zero pads + x->f16 + 4 weight transposes (one launch)
    prep_kernel<<<10336, 256, 0, stream>>>(x, Wq, Wk, Wv, Wo,
                                           xh, wtq, wtk, wtv, wto,
                                           (v4u*)(ws + OFF_KP));

    // 2) fused Q+K+V projections (WMMA, 16x64 strip per wave)
    gemm_qkv<<<3072, 128, 0, stream>>>(xh, wtq, wtk, wtv, bq, bk, bv,
                                       q_h, k_p, v_p);

    // 3) banded attention (WMMA QK^T + softmax + WMMA PV), writes attn output
    attn_wmma<<<1024, 128, 0, stream>>>(q_h, k_p, v_p, ctxh, attn_out);

    // 4) output projection (WMMA) -> fp32 out
    gemm_o<<<1024, 128, 0, stream>>>(ctxh, wto, bo, out);
}
